// DECSeqSelf_41180146434801
// MI455X (gfx1250) — matrix operand order repaired and checked by reading, not compiled, not run on hardware
//
#include <hip/hip_runtime.h>
#include <hip/hip_bf16.h>

typedef _Float16 half_t;
typedef __attribute__((ext_vector_type(16))) _Float16 v16h;
typedef __attribute__((ext_vector_type(8)))  _Float16 v8h;
typedef __attribute__((ext_vector_type(8)))  float    v8f;
typedef __attribute__((ext_vector_type(4)))  unsigned int u32x4;
typedef __attribute__((ext_vector_type(8)))  int      i32x8;
typedef __attribute__((ext_vector_type(4)))  int      i32x4;

#define B_SZ   1024
#define NPTS_C 128
#define KNN_C  5
#define NNODE  (B_SZ * NPTS_C)                     // 131072
#define NFWD   (B_SZ * (NPTS_C - 1))               // 130048
#define NEDGE  (2 * NFWD + NNODE)                  // 391168

static __device__ __forceinline__ float hmaxf(float a, float b) { return a > b ? a : b; }

// ---------------------------------------------------------------------------
// TDM: 2D tile load (data_size = 2 bytes) global -> LDS via Tensor Data Mover.
//   tile: tileD1 rows of tileD0 contiguous f16 elements, row stride = stride0
//   LDS receives the tile packed row-major (tileD0*2 bytes per row).
// ---------------------------------------------------------------------------
static __device__ __forceinline__ void tdm_load_tile_f16(
        unsigned ldsOff, const void* gaddr,
        unsigned tileD0, unsigned tileD1,
        unsigned long long tensorD0, unsigned long long tensorD1,
        unsigned long long stride0) {
    unsigned long long ga = (unsigned long long)(uintptr_t)gaddr;
    u32x4 g0;
    g0[0] = 1u;                                               // count=1 (valid user D#)
    g0[1] = ldsOff;                                           // lds_addr (bytes)
    g0[2] = (unsigned)(ga & 0xFFFFFFFFu);                     // global_addr[31:0]
    g0[3] = (unsigned)((ga >> 32) & 0x01FFFFFFu) | (2u << 30);// global_addr[56:32] | type=2
    i32x8 g1;
    g1[0] = (int)(1u << 16);                                  // wg_mask=0, data_size=1 (2B)
    g1[1] = (int)((tensorD0 & 0xFFFFu) << 16);                // tensor_dim0[15:0] @ [63:48]
    g1[2] = (int)(((tensorD0 >> 16) & 0xFFFFu) |              // tensor_dim0[31:16]
                  ((tensorD1 & 0xFFFFu) << 16));              // tensor_dim1[15:0]
    g1[3] = (int)(((tensorD1 >> 16) & 0xFFFFu) |              // tensor_dim1[31:16]
                  ((tileD0 & 0xFFFFu) << 16));                // tile_dim0 @ [127:112]
    g1[4] = (int)(tileD1 & 0xFFFFu);                          // tile_dim1; tile_dim2=0
    g1[5] = (int)(stride0 & 0xFFFFFFFFu);                     // tensor_dim0_stride[31:0]
    g1[6] = (int)((stride0 >> 32) & 0xFFFFu);                 // stride0[47:32]; stride1=0
    g1[7] = 0;
    i32x4 gz = {0, 0, 0, 0};
#if defined(__clang_major__) && __clang_major__ >= 23
    i32x8 gz8 = {0, 0, 0, 0, 0, 0, 0, 0};
    __builtin_amdgcn_tensor_load_to_lds(g0, g1, gz, gz, gz8, 0);
#else
    __builtin_amdgcn_tensor_load_to_lds(g0, g1, gz, gz, 0);
#endif
}

// low 32 bits of a generic shared-space pointer == workgroup LDS byte offset
static __device__ __forceinline__ unsigned lds_byte_offset(const void* p) {
    return (unsigned)(uintptr_t)p;
}

// ---------------------------------------------------------------------------
// BN fold:  scale = g * rsqrt(rv+eps) ; shift = be - rm*scale
// ---------------------------------------------------------------------------
__global__ void bn_prep_kernel(const float* __restrict__ g, const float* __restrict__ be,
                               const float* __restrict__ rm, const float* __restrict__ rv,
                               float* __restrict__ sc, float* __restrict__ sh, int n) {
    int i = blockIdx.x * blockDim.x + threadIdx.x;
    if (i >= n) return;
    float s = g[i] * rsqrtf(rv[i] + 1e-5f);
    sc[i] = s;
    sh[i] = be[i] - rm[i] * s;
}

// W (Kdim x Nout, f32, row-major)  ->  WT (Nout x Kdim, f16, row-major)
__global__ void transpose_w_kernel(const float* __restrict__ W, half_t* __restrict__ WT,
                                   int Kdim, int Nout) {
    int idx = blockIdx.x * blockDim.x + threadIdx.x;
    if (idx >= Kdim * Nout) return;
    int n = idx / Kdim, k = idx - n * Kdim;
    WT[(size_t)n * Kdim + k] = (half_t)W[(size_t)k * Nout + n];
}

// ---------------------------------------------------------------------------
// conv1 layer 1: per-edge feature build (6) -> 64, relu, BN. Scalar (tiny K).
// ---------------------------------------------------------------------------
__global__ void conv1_l1_kernel(const float* __restrict__ pos, const float* __restrict__ W1,
                                const float* __restrict__ b1, const float* __restrict__ sc,
                                const float* __restrict__ sh, half_t* __restrict__ out) {
    int idx = blockIdx.x * blockDim.x + threadIdx.x;
    if (idx >= NEDGE * 64) return;
    int e = idx >> 6, f = idx & 63;
    int src, dst;
    if (e < NFWD) {
        int b = e / (NPTS_C - 1), j = e % (NPTS_C - 1);
        src = b * NPTS_C + j; dst = src + 1;
    } else if (e < 2 * NFWD) {
        int t = e - NFWD;
        int b = t / (NPTS_C - 1), j = t % (NPTS_C - 1);
        dst = b * NPTS_C + j; src = dst + 1;
    } else {
        src = dst = e - 2 * NFWD;
    }
    float xi0 = pos[dst * 3 + 0], xi1 = pos[dst * 3 + 1], xi2 = pos[dst * 3 + 2];
    float ft[6] = { xi0, xi1, xi2,
                    pos[src * 3 + 0] - xi0, pos[src * 3 + 1] - xi1, pos[src * 3 + 2] - xi2 };
    float v = b1[f];
#pragma unroll
    for (int k = 0; k < 6; ++k) v += ft[k] * W1[k * 64 + f];
    v = hmaxf(v, 0.f);
    v = v * sc[f] + sh[f];
    out[(size_t)e * 64 + f] = (half_t)v;
}

// ---------------------------------------------------------------------------
// WMMA GEMM:  out = BN(relu(A @ W + bias))
//   A:  M x Kdim (f16 row-major)   WT: Nout x Kdim (f16 row-major, pre-transposed)
//   block = 128 threads (4 waves). Tile: 32 rows x 64 cols, K stepped by 32.
//   Each wave: 2 accumulators (rows 0-15 / 16-31 of tile) x its 16-col slice.
//   W tile (64 cols x 32 k) is DMA'd by the Tensor Data Mover into LDS.
//   MODE 0: A from memory. MODE 1 (conv2): A row r = [x1[node], x1[nbr]-x1[node]],
//           node = r/5, nbr = idx[r]; Kdim = 128.
// ---------------------------------------------------------------------------
template <int MODE>
__global__ __launch_bounds__(128)
void gemm_wmma_kernel(const half_t* __restrict__ A, const half_t* __restrict__ WT,
                      const float* __restrict__ bias, const float* __restrict__ scale,
                      const float* __restrict__ shift, half_t* __restrict__ outH,
                      int M, int Kdim, int Nout,
                      const float* __restrict__ xg, const int* __restrict__ nbrIdx) {
    __shared__ half_t ldsA[32 * 32];     // 32 rows x 32 k   (2 KB)
    __shared__ half_t ldsWT[64 * 32];    // [col][k]         (4 KB)

    const int tid   = threadIdx.x;
    const int wave  = tid >> 5;
    const int lane  = tid & 31;
    const int hhalf = lane >> 4;   // 0/1
    const int l16   = lane & 15;
    const int row0  = blockIdx.x * 32;
    const int c0    = blockIdx.y * 64;
    const unsigned wtLds = lds_byte_offset(ldsWT);

    v8f acc0 = {};
    v8f acc1 = {};

    for (int k0 = 0; k0 < Kdim; k0 += 32) {
        __syncthreads();   // protect LDS reuse (frag reads of previous step done)

        // ---- W tile via Tensor Data Mover: 64 rows x 32 halfs from WT[c0+.., k0+..] ----
        if (wave == 0) {
            if (lane == 0) {
                tdm_load_tile_f16(wtLds, WT + (size_t)c0 * Kdim + k0,
                                  /*tileD0=*/32, /*tileD1=*/64,
                                  /*tensorD0=*/(unsigned long long)Kdim,
                                  /*tensorD1=*/(unsigned long long)Nout,
                                  /*stride0=*/(unsigned long long)Kdim);
            }
            __builtin_amdgcn_s_wait_tensorcnt(0);
        }

        // ---- stage A tile: 128 threads, row = t/4, 8-half segment = t%4 ----
        {
            int m = tid >> 2, seg = tid & 3;
            int r = row0 + m;
            if (MODE == 0) {
                const v8h* srcp = (const v8h*)(A + (size_t)r * Kdim + k0 + seg * 8);
                *(v8h*)(ldsA + m * 32 + seg * 8) = *srcp;
                __builtin_prefetch(A + (size_t)r * Kdim + k0 + 32, 0, 1);
            } else {
                int node = r / KNN_C;
                int b    = node >> 7;
                int j    = nbrIdx[r];
                const float* xn = xg + (size_t)node * 64;
                const float* xj = xg + ((size_t)(b << 7) + j) * 64;
                v8h tmp;
#pragma unroll
                for (int i = 0; i < 8; ++i) {
                    int c = k0 + seg * 8 + i;
                    float v = (c < 64) ? xn[c] : (xj[c - 64] - xn[c - 64]);
                    tmp[i] = (half_t)v;
                }
                *(v8h*)(ldsA + m * 32 + seg * 8) = tmp;
            }
        }
        __syncthreads();

        // ---- A fragments (ISA 16-bit A layout: lanes 0-15 K0..7/16..23, 16-31 K8..15/24..31) ----
        v8h a0 = *(const v8h*)(ldsA + l16 * 32 + hhalf * 8);
        v8h a1 = *(const v8h*)(ldsA + l16 * 32 + 16 + hhalf * 8);
        v16h af0 = __builtin_shufflevector(a0, a1, 0, 1, 2, 3, 4, 5, 6, 7,
                                           8, 9, 10, 11, 12, 13, 14, 15);
        v8h a2 = *(const v8h*)(ldsA + (16 + l16) * 32 + hhalf * 8);
        v8h a3 = *(const v8h*)(ldsA + (16 + l16) * 32 + 16 + hhalf * 8);
        v16h af1 = __builtin_shufflevector(a2, a3, 0, 1, 2, 3, 4, 5, 6, 7,
                                           8, 9, 10, 11, 12, 13, 14, 15);
        // ---- B fragment: col = l16 of wave's slice, K = hhalf*8 + e (e<8), 16 + hhalf*8 + (e-8) (e>=8) ----
        const half_t* bcol = ldsWT + (wave * 16 + l16) * 32 + hhalf * 8;
        v8h b0 = *(const v8h*)(bcol);
        v8h b1 = *(const v8h*)(bcol + 16);
        v16h bf = __builtin_shufflevector(b0, b1, 0, 1, 2, 3, 4, 5, 6, 7,
                                          8, 9, 10, 11, 12, 13, 14, 15);

        acc0 = __builtin_amdgcn_wmma_f32_16x16x32_f16(false, af0, false, bf,
                                                      (short)0, acc0, false, false);
        acc1 = __builtin_amdgcn_wmma_f32_16x16x32_f16(false, af1, false, bf,
                                                      (short)0, acc1, false, false);
    }

    // ---- epilogue ----
    const int colg = c0 + wave * 16 + l16;
    const float bv = bias[colg], scv = scale[colg], shv = shift[colg];
#pragma unroll
    for (int i = 0; i < 8; ++i) {
        int row = row0 + i + 8 * hhalf;
        float v = acc0[i] + bv;
        v = hmaxf(v, 0.f);
        v = v * scv + shv;
        outH[(size_t)row * Nout + colg] = (half_t)v;
    }
#pragma unroll
    for (int i = 0; i < 8; ++i) {
        int row = row0 + 16 + i + 8 * hhalf;
        float v = acc1[i] + bv;
        v = hmaxf(v, 0.f);
        v = v * scv + shv;
        outH[(size_t)row * Nout + colg] = (half_t)v;
    }
}

// ---------------------------------------------------------------------------
// segment_max over incoming edges (chain topology: self loop + up to 2 nbrs)
// ---------------------------------------------------------------------------
__global__ void seg_max_kernel(const half_t* __restrict__ h3, float* __restrict__ x1f,
                               half_t* __restrict__ lin_in) {
    int idx = blockIdx.x * blockDim.x + threadIdx.x;
    if (idx >= NNODE * 64) return;
    int node = idx >> 6, f = idx & 63;
    int b = node >> 7, j = node & 127;
    float m = (float)h3[(size_t)(2 * NFWD + node) * 64 + f];               // self loop
    if (j > 0)
        m = hmaxf(m, (float)h3[(size_t)(b * (NPTS_C - 1) + (j - 1)) * 64 + f]);
    if (j < NPTS_C - 1)
        m = hmaxf(m, (float)h3[(size_t)(NFWD + b * (NPTS_C - 1) + j) * 64 + f]);
    x1f[idx] = m;
    lin_in[(size_t)node * 192 + f] = (half_t)m;
}

// ---------------------------------------------------------------------------
// kNN (K=5 smallest squared distance, stable ties -> lower index first)
// ---------------------------------------------------------------------------
__global__ __launch_bounds__(128)
void knn_kernel(const float* __restrict__ x1f, int* __restrict__ nbrIdx) {
    __shared__ float xs[NPTS_C * 64];
    __shared__ float sq[NPTS_C];
    int b = blockIdx.x, t = threadIdx.x;
    const float* xb = x1f + (size_t)b * NPTS_C * 64;
    for (int i = t; i < NPTS_C * 64; i += 128) xs[i] = xb[i];
    __syncthreads();
    float s = 0.f;
#pragma unroll 8
    for (int c = 0; c < 64; ++c) { float v = xs[t * 64 + c]; s += v * v; }
    sq[t] = s;
    __syncthreads();
    float bd[KNN_C]; int bi[KNN_C];
#pragma unroll
    for (int k = 0; k < KNN_C; ++k) { bd[k] = 3.0e38f; bi[k] = 0; }
    for (int j = 0; j < NPTS_C; ++j) {
        float dot = 0.f;
#pragma unroll 8
        for (int c = 0; c < 64; ++c) dot += xs[t * 64 + c] * xs[j * 64 + c];
        float d = sq[t] + sq[j] - 2.f * dot;
        if (d < bd[KNN_C - 1]) {
            int p = KNN_C - 1;
            while (p > 0 && d < bd[p - 1]) { bd[p] = bd[p - 1]; bi[p] = bi[p - 1]; --p; }
            bd[p] = d; bi[p] = j;
        }
    }
#pragma unroll
    for (int k = 0; k < KNN_C; ++k)
        nbrIdx[((size_t)b * NPTS_C + t) * KNN_C + k] = bi[k];
}

// ---------------------------------------------------------------------------
// x2 = max over K neighbors of conv2 output; write into lin_in cols 64..191
// ---------------------------------------------------------------------------
__global__ void x2_max_kernel(const half_t* __restrict__ h2c, half_t* __restrict__ lin_in) {
    int idx = blockIdx.x * blockDim.x + threadIdx.x;
    if (idx >= NNODE * 128) return;
    int node = idx >> 7, f = idx & 127;
    float m = -3.0e38f;
#pragma unroll
    for (int k = 0; k < KNN_C; ++k)
        m = hmaxf(m, (float)h2c[((size_t)node * KNN_C + k) * 128 + f]);
    lin_in[(size_t)node * 192 + 64 + f] = (half_t)m;
}

// ---------------------------------------------------------------------------
// max-pool over NPTS points per batch
// ---------------------------------------------------------------------------
__global__ void pool_max_kernel(const half_t* __restrict__ h, half_t* __restrict__ g, int F) {
    int idx = blockIdx.x * blockDim.x + threadIdx.x;
    if (idx >= B_SZ * F) return;
    int b = idx / F, f = idx - b * F;
    float m = -3.0e38f;
    for (int j = 0; j < NPTS_C; ++j)
        m = hmaxf(m, (float)h[((size_t)b * NPTS_C + j) * F + f]);
    g[idx] = (half_t)m;
}

// ---------------------------------------------------------------------------
// final linear 256 -> 2 (f32 output)
// ---------------------------------------------------------------------------
__global__ void out_kernel(const half_t* __restrict__ gh2, const float* __restrict__ W,
                           const float* __restrict__ bias, float* __restrict__ out) {
    int idx = blockIdx.x * blockDim.x + threadIdx.x;
    if (idx >= B_SZ * 2) return;
    int b = idx >> 1, c = idx & 1;
    float v = bias[c];
    for (int k = 0; k < 256; ++k)
        v += (float)gh2[(size_t)b * 256 + k] * W[k * 2 + c];
    out[idx] = v;
}

// ---------------------------------------------------------------------------
// host launcher
// ---------------------------------------------------------------------------
static inline int cdiv(long long a, long long b) { return (int)((a + b - 1) / b); }

extern "C" void kernel_launch(void* const* d_in, const int* in_sizes, int n_in,
                              void* d_out, int out_size, void* d_ws, size_t ws_size,
                              hipStream_t stream) {
    (void)in_sizes; (void)n_in; (void)out_size; (void)ws_size;

    const float* pos = (const float*)d_in[0];
    const float* c1W[3]  = { (const float*)d_in[3],  (const float*)d_in[9],  (const float*)d_in[15] };
    const float* c1b[3]  = { (const float*)d_in[4],  (const float*)d_in[10], (const float*)d_in[16] };
    const float* c2W  = (const float*)d_in[21]; const float* c2b  = (const float*)d_in[22];
    const float* l1W  = (const float*)d_in[27]; const float* l1b  = (const float*)d_in[28];
    const float* h1W  = (const float*)d_in[33]; const float* h1b  = (const float*)d_in[34];
    const float* h2W  = (const float*)d_in[39]; const float* h2b  = (const float*)d_in[40];
    const float* oW   = (const float*)d_in[45]; const float* ob   = (const float*)d_in[46];

    // ---- carve workspace ----
    char* p = (char*)d_ws;
    auto carve = [&](size_t bytes) { void* r = (void*)p; p += (bytes + 255) & ~(size_t)255; return r; };

    half_t* w2t  = (half_t*)carve(64 * 64 * sizeof(half_t));          // WT: Nout x Kdim
    half_t* w3t  = (half_t*)carve(64 * 64 * sizeof(half_t));
    half_t* wc2t = (half_t*)carve(128 * 128 * sizeof(half_t));
    half_t* wl1t = (half_t*)carve((size_t)1024 * 192 * sizeof(half_t));
    half_t* wh1t = (half_t*)carve((size_t)512 * 1024 * sizeof(half_t));
    half_t* wh2t = (half_t*)carve((size_t)256 * 512 * sizeof(half_t));

    float *sc[7], *sh[7];
    const int bnN[7] = { 64, 64, 64, 128, 1024, 512, 256 };
    for (int i = 0; i < 7; ++i) {
        sc[i] = (float*)carve(bnN[i] * sizeof(float));
        sh[i] = (float*)carve(bnN[i] * sizeof(float));
    }

    half_t* bufA   = (half_t*)carve((size_t)NEDGE * 64 * sizeof(half_t));
    half_t* bufB   = (half_t*)carve((size_t)NEDGE * 64 * sizeof(half_t));
    float*  x1f    = (float*) carve((size_t)NNODE * 64 * sizeof(float));
    half_t* lin_in = (half_t*)carve((size_t)NNODE * 192 * sizeof(half_t));
    int*    nbrIdx = (int*)   carve((size_t)NNODE * KNN_C * sizeof(int));
    half_t* h2c    = (half_t*)carve((size_t)NNODE * KNN_C * 128 * sizeof(half_t));
    half_t* hl1    = (half_t*)carve((size_t)NNODE * 1024 * sizeof(half_t));
    half_t* gbuf   = (half_t*)carve((size_t)B_SZ * 1024 * sizeof(half_t));
    half_t* gh1    = (half_t*)carve((size_t)B_SZ * 512 * sizeof(half_t));
    half_t* gh2    = (half_t*)carve((size_t)B_SZ * 256 * sizeof(half_t));

    // ---- BN folds ----
    const float* bn_g[7]  = { (const float*)d_in[5],  (const float*)d_in[11], (const float*)d_in[17],
                              (const float*)d_in[23], (const float*)d_in[29], (const float*)d_in[35],
                              (const float*)d_in[41] };
    const float* bn_be[7] = { (const float*)d_in[6],  (const float*)d_in[12], (const float*)d_in[18],
                              (const float*)d_in[24], (const float*)d_in[30], (const float*)d_in[36],
                              (const float*)d_in[42] };
    const float* bn_rm[7] = { (const float*)d_in[7],  (const float*)d_in[13], (const float*)d_in[19],
                              (const float*)d_in[25], (const float*)d_in[31], (const float*)d_in[37],
                              (const float*)d_in[43] };
    const float* bn_rv[7] = { (const float*)d_in[8],  (const float*)d_in[14], (const float*)d_in[20],
                              (const float*)d_in[26], (const float*)d_in[32], (const float*)d_in[38],
                              (const float*)d_in[44] };
    for (int i = 0; i < 7; ++i)
        bn_prep_kernel<<<cdiv(bnN[i], 256), 256, 0, stream>>>(bn_g[i], bn_be[i], bn_rm[i], bn_rv[i], sc[i], sh[i], bnN[i]);

    // ---- weight transposes (f32 -> f16, Nout x Kdim) ----
    transpose_w_kernel<<<cdiv(64 * 64, 256), 256, 0, stream>>>(c1W[1], w2t, 64, 64);
    transpose_w_kernel<<<cdiv(64 * 64, 256), 256, 0, stream>>>(c1W[2], w3t, 64, 64);
    transpose_w_kernel<<<cdiv(128 * 128, 256), 256, 0, stream>>>(c2W, wc2t, 128, 128);
    transpose_w_kernel<<<cdiv(192 * 1024, 256), 256, 0, stream>>>(l1W, wl1t, 192, 1024);
    transpose_w_kernel<<<cdiv(1024 * 512, 256), 256, 0, stream>>>(h1W, wh1t, 1024, 512);
    transpose_w_kernel<<<cdiv(512 * 256, 256), 256, 0, stream>>>(h2W, wh2t, 512, 256);

    // ---- conv1 ----
    conv1_l1_kernel<<<cdiv((long long)NEDGE * 64, 256), 256, 0, stream>>>(pos, c1W[0], c1b[0], sc[0], sh[0], bufA);
    gemm_wmma_kernel<0><<<dim3(NEDGE / 32, 1), 128, 0, stream>>>(bufA, w2t, c1b[1], sc[1], sh[1], bufB, NEDGE, 64, 64, nullptr, nullptr);
    gemm_wmma_kernel<0><<<dim3(NEDGE / 32, 1), 128, 0, stream>>>(bufB, w3t, c1b[2], sc[2], sh[2], bufA, NEDGE, 64, 64, nullptr, nullptr);

    // ---- segment max -> x1 ----
    seg_max_kernel<<<cdiv((long long)NNODE * 64, 256), 256, 0, stream>>>(bufA, x1f, lin_in);

    // ---- kNN ----
    knn_kernel<<<B_SZ, 128, 0, stream>>>(x1f, nbrIdx);

    // ---- conv2 (edge-feature gather fused into WMMA A-tile staging) ----
    gemm_wmma_kernel<1><<<dim3(NNODE * KNN_C / 32, 128 / 64), 128, 0, stream>>>(nullptr, wc2t, c2b, sc[3], sh[3], h2c, NNODE * KNN_C, 128, 128, x1f, nbrIdx);

    // ---- max over K -> x2 (cols 64..191 of lin_in) ----
    x2_max_kernel<<<cdiv((long long)NNODE * 128, 256), 256, 0, stream>>>(h2c, lin_in);

    // ---- lin1 (dominant GEMM: 131072 x 192 @ 192 x 1024) ----
    gemm_wmma_kernel<0><<<dim3(NNODE / 32, 1024 / 64), 128, 0, stream>>>(lin_in, wl1t, l1b, sc[4], sh[4], hl1, NNODE, 192, 1024, nullptr, nullptr);

    // ---- global max pool ----
    pool_max_kernel<<<cdiv(B_SZ * 1024, 256), 256, 0, stream>>>(hl1, gbuf, 1024);

    // ---- heads ----
    gemm_wmma_kernel<0><<<dim3(B_SZ / 32, 512 / 64), 128, 0, stream>>>(gbuf, wh1t, h1b, sc[5], sh[5], gh1, B_SZ, 1024, 512, nullptr, nullptr);
    gemm_wmma_kernel<0><<<dim3(B_SZ / 32, 256 / 64), 128, 0, stream>>>(gh1, wh2t, h2b, sc[6], sh[6], gh2, B_SZ, 512, 256, nullptr, nullptr);

    // ---- output projection (f32) ----
    out_kernel<<<cdiv(B_SZ * 2, 256), 256, 0, stream>>>(gh2, oW, ob, (float*)d_out);
}
